// _SelfAttention_37692632989737
// MI455X (gfx1250) — matrix operand order repaired and checked by reading, not compiled, not run on hardware
//
#include <hip/hip_runtime.h>
#include <hip/hip_bf16.h>

typedef __attribute__((ext_vector_type(16))) _Float16 v16h;
typedef __attribute__((ext_vector_type(8)))  _Float16 v8h;
typedef __attribute__((ext_vector_type(8)))  float    v8f;

union V16H { v16h v; v8h h[2]; };

__device__ __forceinline__ v8h ld8h(const _Float16* p) {
    return *reinterpret_cast<const v8h*>(p);
}

__device__ __forceinline__ v8f wmma_f16(v16h a, v16h b, v8f c) {
    // D = A(16x32 f16) * B(32x16 f16) + C(16x16 f32)
    return __builtin_amdgcn_wmma_f32_16x16x32_f16(
        /*neg_a=*/false, a, /*neg_b=*/false, b,
        /*c_mod=*/(short)0, c, /*reuse_a=*/false, /*reuse_b=*/false);
}

// Problem constants
#define B       2
#define C       256
#define S       4096      // h*w = 64*64
#define NH      4
#define HD      64
#define GROUPS  32
#define CH_PER_G 8

// ---------------------------------------------------------------------------
// Kernel 0: fp32 -> f16 weight conversion (runs once per launch, tiny)
// ---------------------------------------------------------------------------
__global__ __launch_bounds__(256) void cvt_f16_kernel(
        const float* __restrict__ src, _Float16* __restrict__ dst, int n) {
    int i = blockIdx.x * 256 + threadIdx.x;
    if (i < n) dst[i] = (_Float16)src[i];
}

// ---------------------------------------------------------------------------
// Kernel 1: GroupNorm statistics. 64 blocks = (b, group); 256 threads each.
// ---------------------------------------------------------------------------
__global__ __launch_bounds__(256) void gn_stats_kernel(
        const float* __restrict__ x, float* __restrict__ stats) {
    int bg = blockIdx.x;                                 // 0..63
    const float* base = x + (size_t)bg * CH_PER_G * S;   // group channels contiguous
    float s = 0.f, ss = 0.f;
    for (int i = threadIdx.x; i < CH_PER_G * S; i += 256) {
        float v = base[i];
        s += v; ss += v * v;
    }
    for (int off = 16; off; off >>= 1) {
        s  += __shfl_xor(s,  off);
        ss += __shfl_xor(ss, off);
    }
    __shared__ float shs[8], shss[8];
    int wave = threadIdx.x >> 5;
    if ((threadIdx.x & 31) == 0) { shs[wave] = s; shss[wave] = ss; }
    __syncthreads();
    if (threadIdx.x == 0) {
        float ts = 0.f, tss = 0.f;
        for (int w = 0; w < 8; w++) { ts += shs[w]; tss += shss[w]; }
        float inv_n = 1.0f / (float)(CH_PER_G * S);
        float mean = ts * inv_n;
        float var  = tss * inv_n - mean * mean;
        stats[bg * 2 + 0] = mean;
        stats[bg * 2 + 1] = rsqrtf(var + 1e-5f);
    }
}

// ---------------------------------------------------------------------------
// Kernel 2: normalize + affine + transpose to normT[b, p, c] (f16, c-contig)
// ---------------------------------------------------------------------------
__global__ __launch_bounds__(256) void gn_apply_kernel(
        const float* __restrict__ x, const float* __restrict__ nw,
        const float* __restrict__ nb, const float* __restrict__ stats,
        _Float16* __restrict__ normT) {
    int idx = blockIdx.x * 256 + threadIdx.x;       // B*S*C threads
    int ch = idx & (C - 1);
    int p  = (idx >> 8) & (S - 1);
    int b  = idx >> 20;
    int sg = (b * GROUPS + (ch >> 3)) * 2;
    float mean = stats[sg], rstd = stats[sg + 1];
    float v = x[((size_t)(b * C + ch)) * S + p];
    normT[idx] = (_Float16)((v - mean) * rstd * nw[ch] + nb[ch]);
}

// ---------------------------------------------------------------------------
// Kernel 3: QKV GEMM.  D[d,p] = sum_c W[d,c] * normT[p,c]  (per batch)
// M=768, N=4096, K=256. Each wave computes a 16x64 tile (4 accumulators),
// so one A operand feeds 4 WMMAs. 8 waves/block -> 768 blocks.
// Scatter: q16[bh,p,hd] (scaled 1/8), k16[bh,p,hd], vT16[bh,hd,p].
// ---------------------------------------------------------------------------
__global__ __launch_bounds__(256) void qkv_gemm_kernel(
        const _Float16* __restrict__ w16, const _Float16* __restrict__ normT,
        _Float16* __restrict__ q16, _Float16* __restrict__ k16,
        _Float16* __restrict__ vT16) {
    int lane = threadIdx.x & 31;
    int wave = threadIdx.x >> 5;
    int tile = blockIdx.x * 8 + wave;        // 6144 wave-tiles
    int batch = tile / 3072;
    int t = tile % 3072;
    int Mbase = (t / 64) << 4;               // 48 m-tiles
    int Nbase = (t & 63) << 6;               // 64 n-supertiles of width 64
    int l15 = lane & 15;
    int hiA = (lane >= 16) ? 16 : 0;
    int hiB = (lane >= 16) ? 16 : 0;

    const _Float16* arow = w16 + (size_t)(Mbase + l15) * C;
    const _Float16* brow = normT + ((size_t)batch * S + Nbase + l15) * C;

    v8f acc0 = {}, acc1 = {}, acc2 = {}, acc3 = {};
    #pragma unroll
    for (int k = 0; k < C; k += 32) {
        V16H a, b0, b1, b2, b3;
        a.h[0] = ld8h(arow + k + hiA);
        a.h[1] = ld8h(arow + k + hiA + 8);
        const _Float16* bp = brow + k + hiB;
        b0.h[0] = ld8h(bp);              b0.h[1] = ld8h(bp + 8);
        b1.h[0] = ld8h(bp + 16 * C);     b1.h[1] = ld8h(bp + 16 * C + 8);
        b2.h[0] = ld8h(bp + 32 * C);     b2.h[1] = ld8h(bp + 32 * C + 8);
        b3.h[0] = ld8h(bp + 48 * C);     b3.h[1] = ld8h(bp + 48 * C + 8);
        acc0 = wmma_f16(a.v, b0.v, acc0);
        acc1 = wmma_f16(a.v, b1.v, acc1);
        acc2 = wmma_f16(a.v, b2.v, acc2);
        acc3 = wmma_f16(a.v, b3.v, acc3);
    }

    int moff = (lane >= 16) ? 8 : 0;
    #pragma unroll
    for (int nt = 0; nt < 4; nt++) {
        v8f acc = (nt == 0) ? acc0 : (nt == 1) ? acc1 : (nt == 2) ? acc2 : acc3;
        int p = Nbase + nt * 16 + l15;
        #pragma unroll
        for (int r = 0; r < 8; r++) {
            int d = Mbase + r + moff;
            float val = acc[r];
            int head = d / 192;
            int rr   = d % 192;
            size_t bh = (size_t)batch * NH + head;
            if (rr < 64)
                q16[(bh * S + p) * HD + rr]          = (_Float16)(val * 0.125f);
            else if (rr < 128)
                k16[(bh * S + p) * HD + (rr - 64)]   = (_Float16)val;
            else
                vT16[(bh * HD + (rr - 128)) * S + p] = (_Float16)val;
        }
    }
}

// ---------------------------------------------------------------------------
// Kernel 4: Flash attention.  512 blocks (bh=8 x 64 p-tiles), 128 threads.
// Each wave owns 16 query rows; iterates keys in chunks of 32.
// Output: attnT[b, p, c] f16 (c-contig) for the projection GEMM.
// ---------------------------------------------------------------------------
__global__ __launch_bounds__(128) void attn_kernel(
        const _Float16* __restrict__ q16, const _Float16* __restrict__ k16,
        const _Float16* __restrict__ vT16, _Float16* __restrict__ attnT) {
    __shared__ _Float16 pstage[4][16 * 32];
    int lane = threadIdx.x & 31;
    int wave = threadIdx.x >> 5;
    int bh    = blockIdx.x >> 6;                 // 0..7
    int pbase = ((blockIdx.x & 63) << 6) + (wave << 4);
    int l15 = lane & 15;
    int hiA = (lane >= 16) ? 8 : 0;
    int hiB = (lane >= 16) ? 16 : 0;

    // Q tile as A operand, K = head_dim (64) split into two 16x32 operands
    const _Float16* qrow = q16 + ((size_t)bh * S + pbase + l15) * HD;
    V16H aq0, aq1;
    aq0.h[0] = ld8h(qrow + hiB);
    aq0.h[1] = ld8h(qrow + hiB + 8);
    aq1.h[0] = ld8h(qrow + 32 + hiB);
    aq1.h[1] = ld8h(qrow + 32 + hiB + 8);

    v8f o0 = {}, o1 = {}, o2 = {}, o3 = {};
    float m8[8], l8[8];
    #pragma unroll
    for (int r = 0; r < 8; r++) { m8[r] = -1e30f; l8[r] = 0.f; }

    const _Float16* kbase = k16  + (size_t)bh * S * HD;
    const _Float16* vbase = vT16 + (size_t)bh * HD * S;
    _Float16* myst = &pstage[wave][0];

    for (int j = 0; j < S; j += 32) {
        // ---- scores: two 16x16 tiles (cols j..j+15 and j+16..j+31) ----
        const _Float16* k0 = kbase + (size_t)(j + l15) * HD + hiB;
        const _Float16* k1 = k0 + 16 * HD;
        V16H bk;
        v8f s0 = {}, s1 = {};
        bk.h[0] = ld8h(k0);      bk.h[1] = ld8h(k0 + 8);
        s0 = wmma_f16(aq0.v, bk.v, s0);
        bk.h[0] = ld8h(k0 + 32); bk.h[1] = ld8h(k0 + 40);
        s0 = wmma_f16(aq1.v, bk.v, s0);
        bk.h[0] = ld8h(k1);      bk.h[1] = ld8h(k1 + 8);
        s1 = wmma_f16(aq0.v, bk.v, s1);
        bk.h[0] = ld8h(k1 + 32); bk.h[1] = ld8h(k1 + 40);
        s1 = wmma_f16(aq1.v, bk.v, s1);

        // ---- online softmax (rows live on 16-lane halves) ----
        #pragma unroll
        for (int r = 0; r < 8; r++) {
            float mx = fmaxf(s0[r], s1[r]);
            mx = fmaxf(mx, __shfl_xor(mx, 1));
            mx = fmaxf(mx, __shfl_xor(mx, 2));
            mx = fmaxf(mx, __shfl_xor(mx, 4));
            mx = fmaxf(mx, __shfl_xor(mx, 8));
            float mnew  = fmaxf(m8[r], mx);
            float alpha = __expf(m8[r] - mnew);
            m8[r] = mnew;
            float p0 = __expf(s0[r] - mnew);
            float p1 = __expf(s1[r] - mnew);
            float rs = p0 + p1;
            rs += __shfl_xor(rs, 1);
            rs += __shfl_xor(rs, 2);
            rs += __shfl_xor(rs, 4);
            rs += __shfl_xor(rs, 8);
            l8[r] = l8[r] * alpha + rs;
            s0[r] = p0; s1[r] = p1;
            o0[r] *= alpha; o1[r] *= alpha; o2[r] *= alpha; o3[r] *= alpha;
        }

        // ---- reshape P: C-layout -> A-layout via per-wave LDS tile ----
        #pragma unroll
        for (int r = 0; r < 8; r++) {
            int m = r + hiA;                 // hiA doubles as C-layout row offset
            myst[m * 32 + l15]      = (_Float16)s0[r];
            myst[m * 32 + 16 + l15] = (_Float16)s1[r];
        }
        V16H ap;
        const _Float16* prow = myst + l15 * 32 + hiB;
        ap.h[0] = ld8h(prow);
        ap.h[1] = ld8h(prow + 8);

        // ---- O(16x64) += P(16x32) x V(32x64): 4 column tiles ----
        V16H bv;
        const _Float16* vc = vbase + (size_t)l15 * S + j + hiB;
        bv.h[0] = ld8h(vc);      bv.h[1] = ld8h(vc + 8);
        o0 = wmma_f16(ap.v, bv.v, o0);
        vc += (size_t)16 * S;
        bv.h[0] = ld8h(vc);      bv.h[1] = ld8h(vc + 8);
        o1 = wmma_f16(ap.v, bv.v, o1);
        vc += (size_t)16 * S;
        bv.h[0] = ld8h(vc);      bv.h[1] = ld8h(vc + 8);
        o2 = wmma_f16(ap.v, bv.v, o2);
        vc += (size_t)16 * S;
        bv.h[0] = ld8h(vc);      bv.h[1] = ld8h(vc + 8);
        o3 = wmma_f16(ap.v, bv.v, o3);
    }

    // ---- epilogue: divide by row sums, store to attnT[b, p, head*64+col] ----
    int b    = bh >> 2;
    int head = bh & 3;
    _Float16* outp = attnT + (size_t)b * S * C + head * HD;
    #pragma unroll
    for (int r = 0; r < 8; r++) {
        int m = r + hiA;
        float inv = 1.0f / l8[r];
        size_t ro = (size_t)(pbase + m) * C;
        outp[ro +  0 + l15] = (_Float16)(o0[r] * inv);
        outp[ro + 16 + l15] = (_Float16)(o1[r] * inv);
        outp[ro + 32 + l15] = (_Float16)(o2[r] * inv);
        outp[ro + 48 + l15] = (_Float16)(o3[r] * inv);
    }
}

// ---------------------------------------------------------------------------
// Kernel 5: output projection + bias + residual.
// D[d,p] = sum_c W2[d,c] * attnT[p,c] + out_b[d] + x[b,d,p]
// Each wave computes 16x64 (4 accumulators). 256 blocks.
// ---------------------------------------------------------------------------
__global__ __launch_bounds__(256) void proj_kernel(
        const _Float16* __restrict__ w16, const float* __restrict__ out_b,
        const float* __restrict__ x, const _Float16* __restrict__ attnT,
        float* __restrict__ out) {
    int lane = threadIdx.x & 31;
    int wave = threadIdx.x >> 5;
    int tile = blockIdx.x * 8 + wave;        // 2048 wave-tiles
    int batch = tile >> 10;
    int t = tile & 1023;
    int Mbase = (t >> 6) << 4;               // 16 m-tiles
    int Nbase = (t & 63) << 6;               // 64 n-supertiles of width 64
    int l15 = lane & 15;
    int hiA = (lane >= 16) ? 16 : 0;
    int hiB = (lane >= 16) ? 16 : 0;

    const _Float16* arow = w16 + (size_t)(Mbase + l15) * C;
    const _Float16* brow = attnT + ((size_t)batch * S + Nbase + l15) * C;

    v8f acc0 = {}, acc1 = {}, acc2 = {}, acc3 = {};
    #pragma unroll
    for (int k = 0; k < C; k += 32) {
        V16H a, b0, b1, b2, b3;
        a.h[0] = ld8h(arow + k + hiA);
        a.h[1] = ld8h(arow + k + hiA + 8);
        const _Float16* bp = brow + k + hiB;
        b0.h[0] = ld8h(bp);              b0.h[1] = ld8h(bp + 8);
        b1.h[0] = ld8h(bp + 16 * C);     b1.h[1] = ld8h(bp + 16 * C + 8);
        b2.h[0] = ld8h(bp + 32 * C);     b2.h[1] = ld8h(bp + 32 * C + 8);
        b3.h[0] = ld8h(bp + 48 * C);     b3.h[1] = ld8h(bp + 48 * C + 8);
        acc0 = wmma_f16(a.v, b0.v, acc0);
        acc1 = wmma_f16(a.v, b1.v, acc1);
        acc2 = wmma_f16(a.v, b2.v, acc2);
        acc3 = wmma_f16(a.v, b3.v, acc3);
    }

    int moff = (lane >= 16) ? 8 : 0;
    #pragma unroll
    for (int nt = 0; nt < 4; nt++) {
        v8f acc = (nt == 0) ? acc0 : (nt == 1) ? acc1 : (nt == 2) ? acc2 : acc3;
        int p = Nbase + nt * 16 + l15;
        #pragma unroll
        for (int r = 0; r < 8; r++) {
            int d = Mbase + r + moff;
            size_t idx = ((size_t)batch * C + d) * S + p;
            out[idx] = acc[r] + out_b[d] + x[idx];
        }
    }
}

// ---------------------------------------------------------------------------
extern "C" void kernel_launch(void* const* d_in, const int* in_sizes, int n_in,
                              void* d_out, int out_size, void* d_ws, size_t ws_size,
                              hipStream_t stream) {
    const float* x      = (const float*)d_in[0];
    const float* norm_w = (const float*)d_in[1];
    const float* norm_b = (const float*)d_in[2];
    const float* qkv_w  = (const float*)d_in[3];
    const float* out_w  = (const float*)d_in[4];
    const float* out_b  = (const float*)d_in[5];
    float* out = (float*)d_out;

    char* ws = (char*)d_ws;
    float* stats = (float*)ws;                         // 64*2 floats
    const size_t HBUF = (size_t)B * S * C;             // 2M halves each
    _Float16* normT   = (_Float16*)(ws + 1024);
    _Float16* q16     = normT + HBUF;
    _Float16* k16     = q16   + HBUF;
    _Float16* vT16    = k16   + HBUF;
    _Float16* attnT   = vT16  + HBUF;
    _Float16* qkv_w16 = attnT + HBUF;                  // 768*256
    _Float16* out_w16 = qkv_w16 + (size_t)3 * C * C;   // 256*256

    cvt_f16_kernel<<<(3 * C * C + 255) / 256, 256, 0, stream>>>(qkv_w, qkv_w16,
                                                                3 * C * C);
    cvt_f16_kernel<<<(C * C + 255) / 256, 256, 0, stream>>>(out_w, out_w16,
                                                            C * C);
    gn_stats_kernel<<<64, 256, 0, stream>>>(x, stats);
    gn_apply_kernel<<<(B * S * C) / 256, 256, 0, stream>>>(x, norm_w, norm_b,
                                                           stats, normT);
    qkv_gemm_kernel<<<768, 256, 0, stream>>>(qkv_w16, normT, q16, k16, vT16);
    attn_kernel<<<512, 128, 0, stream>>>(q16, k16, vT16, attnT);
    proj_kernel<<<256, 256, 0, stream>>>(out_w16, out_b, x, attnT, out);
}
